// KAVNN_GO_14293651161791
// MI455X (gfx1250) — matrix operand order repaired and checked by reading, not compiled, not run on hardware
//
#include <hip/hip_runtime.h>
#include <hip/hip_bf16.h>
#include <math.h>

typedef __attribute__((ext_vector_type(16))) _Float16 v16h;
typedef __attribute__((ext_vector_type(8)))  _Float16 v8h;
typedef __attribute__((ext_vector_type(8)))  float    v8f;

#define B_SZ 256
#define NGO  8000
#define NKE  2000
#define DD   16
#define DRUG 2048
#define XCOLS (NGO + DRUG)

// Branchless tanh: 1 - 2 * rcp(exp2(2*log2e*x) + 1).
// exp2 -> v_exp_f32 (TRANS), rcp -> v_rcp_f32; exact at +/-inf, no EXEC branches.
__device__ __forceinline__ float fast_tanh(float x) {
  float e = __builtin_amdgcn_exp2f(x * 2.8853900817779268f);
  float r = __builtin_amdgcn_rcpf(e + 1.0f);
  return fmaf(-2.0f, r, 1.0f);
}

// ---------------------------------------------------------------- zero
__global__ __launch_bounds__(256) void zero_f32(float* __restrict__ p, long n4) {
  long i = (long)blockIdx.x * blockDim.x + threadIdx.x;
  if (i < n4) ((float4*)p)[i] = make_float4(0.f, 0.f, 0.f, 0.f);
}

// ---------------------------------------------------------------- go -> ke scatter
// One block per edge (uniform src/dst -> scalar loads), 256 threads = batch dim.
// ke[b,dst,d] += tanh(x[b,src] * W_dec[src,d] + b_dec[src,d])
__global__ __launch_bounds__(256) void go_scatter(
    const float* __restrict__ x, const int* __restrict__ src, const int* __restrict__ dst,
    const float* __restrict__ Wdec, const float* __restrict__ bdec,
    float* __restrict__ ke, int nE)
{
  int e = blockIdx.x;
  if (e >= nE) return;
  int g = src[e];
  int k = dst[e];
  float w[DD], bb[DD];
#pragma unroll
  for (int d = 0; d < DD; ++d) { w[d] = Wdec[g * DD + d]; bb[d] = bdec[g * DD + d]; }
  int b = threadIdx.x;
  float xv = x[(long)b * XCOLS + g];
  float* o = ke + ((long)b * NKE + k) * DD;
#pragma unroll
  for (int d = 0; d < DD; ++d)
    atomicAdd(o + d, fast_tanh(fmaf(xv, w[d], bb[d])));
}

// ---------------------------------------------------------------- ke -> ke scatter (agg)
__global__ __launch_bounds__(256) void ke_scatter(
    const float* __restrict__ ke, const int* __restrict__ src, const int* __restrict__ dst,
    float* __restrict__ agg, int nE)
{
  int e = blockIdx.x;
  if (e >= nE) return;
  int s = src[e];
  int t = dst[e];
  int b = threadIdx.x;
  const float* ip = ke + ((long)b * NKE + s) * DD;
  float* op = agg + ((long)b * NKE + t) * DD;
  float4 v0 = *(const float4*)(ip + 0);
  float4 v1 = *(const float4*)(ip + 4);
  float4 v2 = *(const float4*)(ip + 8);
  float4 v3 = *(const float4*)(ip + 12);
  atomicAdd(op + 0,  v0.x); atomicAdd(op + 1,  v0.y); atomicAdd(op + 2,  v0.z); atomicAdd(op + 3,  v0.w);
  atomicAdd(op + 4,  v1.x); atomicAdd(op + 5,  v1.y); atomicAdd(op + 6,  v1.z); atomicAdd(op + 7,  v1.w);
  atomicAdd(op + 8,  v2.x); atomicAdd(op + 9,  v2.y); atomicAdd(op + 10, v2.z); atomicAdd(op + 11, v2.w);
  atomicAdd(op + 12, v3.x); atomicAdd(op + 13, v3.y); atomicAdd(op + 14, v3.z); atomicAdd(op + 15, v3.w);
}

// ---------------------------------------------------------------- row-wise 16x16 GEMM + tanh (WMMA)
// out[row,:] = tanh((in[row,:] (+agg[row,:])) @ W16 + bias), K=16 padded to 32.
// One wave per 16-row tile; safe in place (full tile read before write).
__global__ __launch_bounds__(256) void ke_gemm16(
    const float* __restrict__ in, const float* __restrict__ agg,
    const float* __restrict__ W16, const float* __restrict__ bias,
    float* __restrict__ out, int ntiles)
{
  int wave = blockIdx.x * (blockDim.x >> 5) + (threadIdx.x >> 5);
  if (wave >= ntiles) return;                 // wave-uniform
  int lane = threadIdx.x & 31;
  int m  = lane & 15;                         // A: M index, C/D & B: N index
  int hi = lane >> 4;

  long row = (long)wave * 16 + m;
  const float* ap = in + row * DD + hi * 8;   // lanes0-15: K 0..7 ; lanes16-31: K 8..15
  float4 q0 = *(const float4*)(ap + 0);
  float4 q1 = *(const float4*)(ap + 4);
  if (agg) {
    const float* gp = agg + row * DD + hi * 8;
    float4 g0 = *(const float4*)(gp + 0);
    float4 g1 = *(const float4*)(gp + 4);
    q0.x += g0.x; q0.y += g0.y; q0.z += g0.z; q0.w += g0.w;
    q1.x += g1.x; q1.y += g1.y; q1.z += g1.z; q1.w += g1.w;
  }
  v16h a;
  a[0] = (_Float16)q0.x; a[1] = (_Float16)q0.y; a[2] = (_Float16)q0.z; a[3] = (_Float16)q0.w;
  a[4] = (_Float16)q1.x; a[5] = (_Float16)q1.y; a[6] = (_Float16)q1.z; a[7] = (_Float16)q1.w;
#pragma unroll
  for (int h = 8; h < 16; ++h) a[h] = (_Float16)0.f;   // K 16..31 pad

  // B: elements 0..7 hold K=8*hi+h of column n; elements 8..15 are the K-pad (zero).
  // Load unconditionally (no EXEC divergence), select with cndmask.
  v16h bv;
#pragma unroll
  for (int h = 0; h < 16; ++h) {
    float wv = W16[((h & 7) + 8 * hi) * 16 + m];
    bv[h] = (h >= 8) ? (_Float16)0.f : (_Float16)wv;
  }

  v8f c = {0.f, 0.f, 0.f, 0.f, 0.f, 0.f, 0.f, 0.f};
  c = __builtin_amdgcn_wmma_f32_16x16x32_f16(false, a, false, bv, (short)0, c, false, false);

  float bn = bias[m];
  long base = (long)wave * 16;
#pragma unroll
  for (int r = 0; r < 8; ++r)
    out[(base + r + 8 * hi) * DD + m] = fast_tanh(c[r] + bn);
}

// ---------------------------------------------------------------- ke . w_ke reduction
__global__ __launch_bounds__(256) void ke_reduce(
    const float* __restrict__ ke, const float* __restrict__ wke,
    const float* __restrict__ bke, float* __restrict__ kes, long nrows)
{
  long i = (long)blockIdx.x * blockDim.x + threadIdx.x;
  if (i >= nrows) return;
  const float* r = ke + i * DD;
  float s = 0.f;
#pragma unroll
  for (int d = 0; d < DD; ++d) s = fmaf(r[d], wke[d], s);
  kes[i] = fast_tanh(s + bke[0]);
}

// ---------------------------------------------------------------- tissue gather
__global__ __launch_bounds__(256) void gather_tissue(
    const float* __restrict__ kes, const int* __restrict__ tissue,
    float* __restrict__ bio0, int nt)
{
  int i = blockIdx.x * blockDim.x + threadIdx.x;
  if (i >= B_SZ * nt) return;
  int b = i / nt, t = i - b * nt;
  bio0[i] = kes[(long)b * NKE + tissue[t]];
}

// ---------------------------------------------------------------- generic WMMA GEMM + bias + ReLU
// C = relu(A(MxK, row stride lda) @ W(KxN) + bias). Block = 8 waves sharing
// two 16x32 f16 A-tiles in LDS per barrier pair; wave w owns the N-strip
// [blockIdx.y*128 + w*16, +16). K % 64 == 0. Two independent accumulators so
// the two v_wmma per iteration have no RAW dependency (pipeline back-to-back).
__global__ __launch_bounds__(256) void gemm_relu_wmma(
    const float* __restrict__ A, int lda,
    const float* __restrict__ W, const float* __restrict__ bias,
    float* __restrict__ out, int N, int K)
{
  __shared__ __align__(16) _Float16 As[2][16][32];
  int tid  = threadIdx.x;
  int wave = tid >> 5, lane = tid & 31;
  int m0 = blockIdx.x * 16;
  int n0 = blockIdx.y * 128 + wave * 16;
  int m  = lane & 15;
  int hi = lane >> 4;
  int n  = n0 + m;

  v8f c0 = {0.f, 0.f, 0.f, 0.f, 0.f, 0.f, 0.f, 0.f};
  v8f c1 = {0.f, 0.f, 0.f, 0.f, 0.f, 0.f, 0.f, 0.f};

  for (int kb = 0; kb < K; kb += 64) {
    // stage two 16x32 A tiles (kb, kb+32) as f16 in LDS: 1024 halves, 4/thread
    for (int i = tid; i < 1024; i += 256) {
      int buf = i >> 9;
      int r   = (i >> 5) & 15;
      int cc  = i & 31;
      As[buf][r][cc] = (_Float16)A[(long)(m0 + r) * lda + kb + (buf << 5) + cc];
    }
    __syncthreads();

    int kbb = kb + hi * 8;                      // B: K=kb+8*hi+h (h<8), kb+16+8*hi+(h-8) (h>=8)
    v16h bv0, bv1;
#pragma unroll
    for (int h = 0; h < 16; ++h) {
      bv0[h] = (_Float16)W[(long)(kbb + h + (h & 8)) * N + n];
      bv1[h] = (_Float16)W[(long)(kbb + 32 + h + (h & 8)) * N + n];
    }

    v8h lo0 = *(const v8h*)&As[0][m][hi * 8];
    v8h hh0 = *(const v8h*)&As[0][m][16 + hi * 8];
    v16h a0 = __builtin_shufflevector(lo0, hh0, 0, 1, 2, 3, 4, 5, 6, 7,
                                                8, 9, 10, 11, 12, 13, 14, 15);
    v8h lo1 = *(const v8h*)&As[1][m][hi * 8];
    v8h hh1 = *(const v8h*)&As[1][m][16 + hi * 8];
    v16h a1 = __builtin_shufflevector(lo1, hh1, 0, 1, 2, 3, 4, 5, 6, 7,
                                                8, 9, 10, 11, 12, 13, 14, 15);

    c0 = __builtin_amdgcn_wmma_f32_16x16x32_f16(false, a0, false, bv0, (short)0, c0, false, false);
    c1 = __builtin_amdgcn_wmma_f32_16x16x32_f16(false, a1, false, bv1, (short)0, c1, false, false);
    __syncthreads();
  }

  float bn = bias[n];
#pragma unroll
  for (int r = 0; r < 8; ++r) {
    float v = c0[r] + c1[r] + bn;
    out[(long)(m0 + r + 8 * hi) * N + n] = v > 0.f ? v : 0.f;
  }
}

// ---------------------------------------------------------------- final prediction head
__global__ __launch_bounds__(256) void pred_kernel(
    const float* __restrict__ bio2, const float* __restrict__ drug2,
    const float* __restrict__ Wp, const float* __restrict__ bp, float* __restrict__ out)
{
  int b = threadIdx.x;
  float s = bp[0];
#pragma unroll 4
  for (int i = 0; i < 128; ++i) s = fmaf(bio2[b * 128 + i], Wp[i], s);
#pragma unroll 4
  for (int i = 0; i < 128; ++i) s = fmaf(drug2[b * 128 + i], Wp[128 + i], s);
  out[b] = s;
}

// ----------------------------------------------------------------
extern "C" void kernel_launch(void* const* d_in, const int* in_sizes, int n_in,
                              void* d_out, int out_size, void* d_ws, size_t ws_size,
                              hipStream_t stream) {
  const float* x       = (const float*)d_in[0];
  const int*   go_src  = (const int*)d_in[1];
  const int*   go_dst  = (const int*)d_in[2];
  const int*   kk_src  = (const int*)d_in[3];
  const int*   kk_dst  = (const int*)d_in[4];
  const int*   tissue  = (const int*)d_in[5];
  const float* W_dec   = (const float*)d_in[6];
  const float* b_dec   = (const float*)d_in[7];
  const float* W_g2k   = (const float*)d_in[8];
  const float* b_g2k   = (const float*)d_in[9];
  const float* W_kk    = (const float*)d_in[10];
  const float* b_kk    = (const float*)d_in[11];
  const float* w_ke    = (const float*)d_in[12];
  const float* b_ke    = (const float*)d_in[13];
  const float* W_bio1  = (const float*)d_in[14];
  const float* b_bio1  = (const float*)d_in[15];
  const float* W_bio2  = (const float*)d_in[16];
  const float* b_bio2  = (const float*)d_in[17];
  const float* W_drug1 = (const float*)d_in[18];
  const float* b_drug1 = (const float*)d_in[19];
  const float* W_drug2 = (const float*)d_in[20];
  const float* b_drug2 = (const float*)d_in[21];
  const float* W_pred  = (const float*)d_in[22];
  const float* b_pred  = (const float*)d_in[23];
  float* out = (float*)d_out;

  int nE_go = in_sizes[1];
  int nE_kk = in_sizes[3];
  int nT    = in_sizes[5];

  float* ws = (float*)d_ws;
  size_t off = 0;
  float* KE   = ws + off; off += (size_t)B_SZ * NKE * DD;   // 32.8 MB
  float* AGG  = ws + off; off += (size_t)B_SZ * NKE * DD;   // 32.8 MB
  float* KES  = ws + off; off += (size_t)B_SZ * NKE;
  float* BIO0 = ws + off; off += (size_t)B_SZ * 512;
  float* BIO1 = ws + off; off += (size_t)B_SZ * 256;
  float* BIO2 = ws + off; off += (size_t)B_SZ * 128;
  float* DR1  = ws + off; off += (size_t)B_SZ * 512;
  float* DR2  = ws + off; off += (size_t)B_SZ * 128;
  (void)ws_size; (void)n_in; (void)out_size;

  long keN4 = (long)B_SZ * NKE * DD / 4;
  int zb = (int)((keN4 + 255) / 256);

  zero_f32<<<zb, 256, 0, stream>>>(KE, keN4);
  go_scatter<<<nE_go, 256, 0, stream>>>(x, go_src, go_dst, W_dec, b_dec, KE, nE_go);

  int ntiles = B_SZ * NKE / 16;              // 32000 full waves
  int g16b = (ntiles + 7) / 8;
  ke_gemm16<<<g16b, 256, 0, stream>>>(KE, nullptr, W_g2k, b_g2k, KE, ntiles);

  for (int l = 0; l < 3; ++l) {
    zero_f32<<<zb, 256, 0, stream>>>(AGG, keN4);
    ke_scatter<<<nE_kk, 256, 0, stream>>>(KE, kk_src, kk_dst, AGG, nE_kk);
    ke_gemm16<<<g16b, 256, 0, stream>>>(KE, AGG, W_kk + l * DD * DD, b_kk + l * DD, KE, ntiles);
  }

  long nrows = (long)B_SZ * NKE;
  ke_reduce<<<(int)((nrows + 255) / 256), 256, 0, stream>>>(KE, w_ke, b_ke, KES, nrows);
  gather_tissue<<<(B_SZ * nT + 255) / 256, 256, 0, stream>>>(KES, tissue, BIO0, nT);

  gemm_relu_wmma<<<dim3(B_SZ / 16, 2), 256, 0, stream>>>(BIO0, 512, W_bio1, b_bio1, BIO1, 256, 512);
  gemm_relu_wmma<<<dim3(B_SZ / 16, 1), 256, 0, stream>>>(BIO1, 256, W_bio2, b_bio2, BIO2, 128, 256);
  gemm_relu_wmma<<<dim3(B_SZ / 16, 4), 256, 0, stream>>>(x + NGO, XCOLS, W_drug1, b_drug1, DR1, 512, 2048);
  gemm_relu_wmma<<<dim3(B_SZ / 16, 1), 256, 0, stream>>>(DR1, 512, W_drug2, b_drug2, DR2, 128, 512);

  pred_kernel<<<1, 256, 0, stream>>>(BIO2, DR2, W_pred, b_pred, out);
}
